// ClassficationAndMDCA_64330020159688
// MI455X (gfx1250) — compile-verified, matrix operand order checked by reading, not run on hardware
//
#include <hip/hip_runtime.h>
#include <hip/hip_bf16.h>
#include <math.h>

#define B_ROWS 8192
#define C_CLS  32000
#define RBLK   64                       // row blocks in conf pass
#define ROWS_PER_BLK (B_ROWS / RBLK)    // 128
#define DEPTH  8                        // async-to-LDS pipeline depth

// workspace byte offsets (total ~8.2 MB)
#define LSE_OFF 0u           // 8192 f32
#define CE_OFF  32768u       // 8192 f32
#define CNT_OFF 65536u       // 32000 i32
#define CP_OFF  193536u      // 64 x 32000 f32 partial conf sums
#define CS_OFF  8385536u     // 32000 f32 conf sums

typedef __attribute__((ext_vector_type(16))) _Float16 v16h;
typedef __attribute__((ext_vector_type(8)))  float    v8f;

// ---- CDNA5 async global->LDS copy (ASYNCcnt path) ----------------------
__device__ __forceinline__ void async_copy_f32(const float* gsrc, unsigned lds_off) {
  // GLOBAL_LOAD_ASYNC_TO_LDS_B32, GV mode: vdst = LDS byte offset, vaddr = 64-bit global addr
  asm volatile("global_load_async_to_lds_b32 %0, %1, off"
               :: "v"(lds_off), "v"(gsrc) : "memory");
}
__device__ __forceinline__ void wait_async_le7() { asm volatile("s_wait_asynccnt 7" ::: "memory"); }
__device__ __forceinline__ void wait_async_0()  { asm volatile("s_wait_asynccnt 0" ::: "memory"); }

// ---- K0: zero class counts ---------------------------------------------
__global__ __launch_bounds__(256) void k_zero_counts(int* __restrict__ counts) {
  int i = blockIdx.x * 256 + threadIdx.x;
  if (i < C_CLS) counts[i] = 0;
}

// ---- K1: integer histogram of targets (deterministic int atomics) ------
__global__ __launch_bounds__(256) void k_hist(const int* __restrict__ targets,
                                              int* __restrict__ counts) {
  int i = blockIdx.x * 256 + threadIdx.x;
  if (i < B_ROWS) atomicAdd(&counts[targets[i]], 1);
}

// ---- K2: per-row online softmax stats (single read of the row) ---------
__global__ __launch_bounds__(256) void k_rowstats(const float* __restrict__ logits,
                                                  const int* __restrict__ targets,
                                                  float* __restrict__ lse_out,
                                                  float* __restrict__ ce_out) {
  const int row = blockIdx.x;
  const int tid = threadIdx.x;
  const float4* rp = (const float4*)(logits + (size_t)row * C_CLS);

  float m = -INFINITY, s = 0.f;
  auto upd = [&](float x) {
    if (x > m) { s = s * __expf(m - x) + 1.f; m = x; }
    else       { s += __expf(x - m); }
  };
#pragma unroll 4
  for (int k = 0; k < 31; ++k) {               // 31*256 = 7936 float4s
    float4 v = rp[tid + (k << 8)];
    upd(v.x); upd(v.y); upd(v.z); upd(v.w);
  }
  if (tid < 64) {                              // tail: 64 float4s -> 8000 total
    float4 v = rp[7936 + tid];
    upd(v.x); upd(v.y); upd(v.z); upd(v.w);
  }

  __shared__ float ms[256], ss[256];
  ms[tid] = m; ss[tid] = s;
  __syncthreads();
  for (int off = 128; off > 0; off >>= 1) {
    if (tid < off) {
      float m2 = ms[tid + off], s2 = ss[tid + off];
      float m1 = ms[tid],       s1 = ss[tid];
      float mm = fmaxf(m1, m2);
      ss[tid] = s1 * __expf(m1 - mm) + s2 * __expf(m2 - mm);
      ms[tid] = mm;
    }
    __syncthreads();
  }
  if (tid == 0) {
    float lse = ms[0] + __logf(ss[0]);
    lse_out[row] = lse;
    int t = targets[row];
    ce_out[row] = lse - logits[(size_t)row * C_CLS + t];   // -log p[target]
  }
}

// ---- K3: per-class prob accumulation with 8-deep async-to-LDS ring -----
__global__ __launch_bounds__(256) void k_confpart(const float* __restrict__ logits,
                                                  const float* __restrict__ lse,
                                                  float* __restrict__ conf_partial) {
  const int tid  = threadIdx.x;
  const int c    = blockIdx.x * 256 + tid;      // class owned by this thread
  const int rb   = blockIdx.y;
  const int row0 = rb * ROWS_PER_BLK;

  __shared__ float buf[DEPTH][256];
  __shared__ float lsb[ROWS_PER_BLK];
  if (tid < ROWS_PER_BLK) lsb[tid] = lse[row0 + tid];
  __syncthreads();

  const float* p = logits + (size_t)row0 * C_CLS + c;

  // prime the pipeline: rows 0..DEPTH-1
#pragma unroll
  for (int i = 0; i < DEPTH; ++i)
    async_copy_f32(p + (size_t)i * C_CLS, (unsigned)(size_t)&buf[i][tid]);

  float acc = 0.f;
  int r = 0;
  for (; r < ROWS_PER_BLK - DEPTH; ++r) {
    wait_async_le7();                           // oldest in-flight load done (in-order)
    int slot = r & (DEPTH - 1);
    float x = buf[slot][tid];
    acc += __expf(x - lsb[r]);
    async_copy_f32(p + (size_t)(r + DEPTH) * C_CLS, (unsigned)(size_t)&buf[slot][tid]);
  }
  wait_async_0();                               // drain
  for (; r < ROWS_PER_BLK; ++r) {
    float x = buf[r & (DEPTH - 1)][tid];
    acc += __expf(x - lsb[r]);
  }
  conf_partial[(size_t)rb * C_CLS + c] = acc;   // no atomics: deterministic
}

// ---- K4: WMMA reduction of row-block partials --------------------------
// conf_sum[c] = sum_y conf_partial[y][c] is ones[1x64] x P[64x32000]:
// with A == all-ones, D[m,n] = sum_k B[k,n], which is permutation-invariant
// in K, so each lane just supplies 16 distinct y-partials for its column
// (lane halves cover complementary y-ranges). Two chained K=32 WMMAs
// reduce all 64 partials for 16 classes per wave.
__global__ __launch_bounds__(256) void k_reduce_partials_wmma(const float* __restrict__ cp,
                                                              float* __restrict__ conf_sum) {
  const int tid     = threadIdx.x;
  const int lane    = tid & 31;
  const int wave    = tid >> 5;
  const int halfsel = lane >> 4;                 // which 16-row set this lane covers
  const int n       = lane & 15;                 // column within the 16-class tile
  const int c       = blockIdx.x * 128 + wave * 16 + n;   // 8 waves * 16 classes per block

  v16h a;
#pragma unroll
  for (int j = 0; j < 16; ++j) a[j] = (_Float16)1.0f;     // ones A matrix

  v8f acc = {};
#pragma unroll
  for (int step = 0; step < 2; ++step) {
    const int ybase = step * 32 + halfsel * 16;
    v16h b;
#pragma unroll
    for (int j = 0; j < 16; ++j)
      b[j] = (_Float16)cp[(size_t)(ybase + j) * C_CLS + c];
    // 8 args: (neg_a, A, neg_b, B, c_mod, C, reuse_a, reuse_b)
    acc = __builtin_amdgcn_wmma_f32_16x16x32_f16(false, a, false, b,
                                                 (short)0, acc, false, false);
  }
  // D VGPR0: lanes 0-15 hold row M=0 (all rows identical) -> class sums
  if (lane < 16) conf_sum[c] = acc[0];
}

// ---- K5: finalize scalar loss ------------------------------------------
__global__ __launch_bounds__(256) void k_finalize(const float* __restrict__ cepart,
                                                  const float* __restrict__ conf_sum,
                                                  const int* __restrict__ counts,
                                                  float* __restrict__ out) {
  int tid = threadIdx.x;
  float ce = 0.f;
  for (int i = tid; i < B_ROWS; i += 256) ce += cepart[i];
  float md = 0.f;
  for (int c = tid; c < C_CLS; c += 256) md += fabsf(conf_sum[c] - (float)counts[c]);

  __shared__ float r1[256], r2[256];
  r1[tid] = ce; r2[tid] = md;
  __syncthreads();
  for (int off = 128; off > 0; off >>= 1) {
    if (tid < off) { r1[tid] += r1[tid + off]; r2[tid] += r2[tid + off]; }
    __syncthreads();
  }
  if (tid == 0)
    out[0] = r1[0] / (float)B_ROWS + r2[0] / ((float)B_ROWS * (float)C_CLS);
}

extern "C" void kernel_launch(void* const* d_in, const int* in_sizes, int n_in,
                              void* d_out, int out_size, void* d_ws, size_t ws_size,
                              hipStream_t stream) {
  const float* logits  = (const float*)d_in[0];
  const int*   targets = (const int*)d_in[1];
  char* ws = (char*)d_ws;
  float* lse   = (float*)(ws + LSE_OFF);
  float* cep   = (float*)(ws + CE_OFF);
  int*   cnt   = (int*)(ws + CNT_OFF);
  float* cpart = (float*)(ws + CP_OFF);
  float* csum  = (float*)(ws + CS_OFF);
  float* out   = (float*)d_out;

  k_zero_counts<<<(C_CLS + 255) / 256, 256, 0, stream>>>(cnt);
  k_hist<<<(B_ROWS + 255) / 256, 256, 0, stream>>>(targets, cnt);
  k_rowstats<<<B_ROWS, 256, 0, stream>>>(logits, targets, lse, cep);
  k_confpart<<<dim3(C_CLS / 256, RBLK), 256, 0, stream>>>(logits, lse, cpart);
  k_reduce_partials_wmma<<<C_CLS / 128, 256, 0, stream>>>(cpart, csum);
  k_finalize<<<1, 256, 0, stream>>>(cep, csum, cnt, out);
}
